// LocationSensitiveAttentionLayer_48670569398980
// MI455X (gfx1250) — matrix operand order repaired and checked by reading, not compiled, not run on hardware
//
#include <hip/hip_runtime.h>
#include <hip/hip_bf16.h>

// Problem constants (from setup_inputs)
#define B_   16
#define TE   1024
#define TD   512
#define DD   512
#define FF   32
#define KN   31
#define PADC 15
#define ROWS 64   // encoder rows per block in the step kernel

typedef __attribute__((ext_vector_type(16))) __bf16 v16bf;
typedef __attribute__((ext_vector_type(8)))  float  v8f;

// ---------- CDNA5 async-to-LDS support (guarded; falls back to plain copies) ----------

#if defined(__AMDGCN__) && __has_builtin(__builtin_amdgcn_global_load_async_to_lds_b128)
#define ASYNC_LDS 1
// Builtin signature (from hipcc diagnostic): (v4i32 AS1*, v4i32 AS3*, imm offset, imm cpol)
typedef __attribute__((__vector_size__(16))) int v4i_;
typedef __attribute__((address_space(1))) v4i_ as1_v4i;
typedef __attribute__((address_space(3))) v4i_ as3_v4i;
__device__ __forceinline__ void async_copy_b128(const void* g, void* l) {
    // GLOBAL_LOAD_ASYNC_TO_LDS_B128: per-lane global->LDS DMA, tracked by ASYNCcnt
    __builtin_amdgcn_global_load_async_to_lds_b128(
        (as1_v4i*)g, (as3_v4i*)l, 0, 0);
}
__device__ __forceinline__ void wait_async0() {
#if __has_builtin(__builtin_amdgcn_s_wait_asynccnt)
    __builtin_amdgcn_s_wait_asynccnt(0);
#else
    asm volatile("s_wait_asynccnt 0x0" ::: "memory");
#endif
}
#else
#define ASYNC_LDS 0
#endif

// ---------- helpers ----------

__device__ __forceinline__ float fast_tanh(float x) {
#if __has_builtin(__builtin_amdgcn_tanhf)
    return __builtin_amdgcn_tanhf(x);   // v_tanh_f32 (gfx1250 TRANS op)
#else
    return tanhf(x);
#endif
}

struct U128x2 { uint4 a, b; };

// Load 16 contiguous bf16 (two 16B chunks) -> v16bf. Used for A fragments:
// lane half 0 holds K=0..15, half 1 holds K=16..31 (contiguous in our staging).
__device__ __forceinline__ v16bf load16bf_contig(const __bf16* p) {
    U128x2 u;
    u.a = *(const uint4*)(p);
    u.b = *(const uint4*)(p + 8);
    return __builtin_bit_cast(v16bf, u);
}

// Load the 16-bit A-matrix lane pattern: elements 0..7 = K base+0..7,
// elements 8..15 = K base+16..23 (base already includes the +8 for lane half 1).
__device__ __forceinline__ v16bf load16bf_split(const __bf16* p) {
    U128x2 u;
    u.a = *(const uint4*)(p);
    u.b = *(const uint4*)(p + 16);
    return __builtin_bit_cast(v16bf, u);
}

// ---------- K0: one-time conversions + zero the cumulative alignment ----------

__global__ void k_init(const float* __restrict__ enc, const float* __restrict__ w_loc,
                       __bf16* __restrict__ encb, __bf16* __restrict__ wlocT,
                       float* __restrict__ cum) {
    const long stride = (long)gridDim.x * blockDim.x;
    const long i0 = (long)blockIdx.x * blockDim.x + threadIdx.x;
    const long n_enc = (long)B_ * TE * DD;
    for (long i = i0; i < n_enc; i += stride) encb[i] = (__bf16)enc[i];
    for (long i = i0; i < FF * DD; i += stride) {
        int k = (int)(i / DD), d = (int)(i % DD);
        wlocT[d * FF + k] = (__bf16)w_loc[i];   // transpose: [D][F]
    }
    for (long i = i0; i < B_ * TE; i += stride) cum[i] = 0.f;
}

// ---------- K1: per-step conv -> loc GEMM (WMMA) -> fused tanh reduction ----------
// grid (TE/ROWS, B_), 128 threads (4 waves). Wave w owns M-tile w of the 64-row tile.

__launch_bounds__(128)
__global__ void k_step(const float* __restrict__ enc,
                       const float* __restrict__ dec,
                       const float* __restrict__ conv_w,   // [KN][1][FF]
                       const float* __restrict__ conv_b,   // [FF]
                       const __bf16* __restrict__ wlocT,   // [DD][FF]
                       const float* __restrict__ v_a,
                       const float* __restrict__ b_a,
                       const float* __restrict__ cum,      // prev_align (cumulative)
                       float* __restrict__ e_scr,          // [B_][TE]
                       int t) {
    __shared__ float  xh[ROWS + KN - 1];
    __shared__ float  cw[KN][FF];
    __shared__ __align__(16) __bf16 fT[ROWS][FF + 8];      // A tile (padded rows)
    __shared__ float  qvb[DD];                              // query + b_a
    __shared__ float  va[DD];
    __shared__ __align__(16) __bf16 wl[DD][FF + 8];         // w_loc^T (padded rows)

    const int tid  = threadIdx.x;
    const int b    = blockIdx.y;
    const int s0   = blockIdx.x * ROWS;
    const int lane = tid & 31;
    const int wv   = tid >> 5;

    // Stage w_loc^T into LDS: 512 rows x 4 uint4 (32 bf16) each.
    {
        const uint4* src = (const uint4*)wlocT;
        for (int i = tid; i < DD * 4; i += 128) {
            int row = i >> 2, q = i & 3;
#if ASYNC_LDS
            async_copy_b128(src + i, &wl[row][q * 8]);       // ASYNCcnt path
#else
            *(uint4*)&wl[row][q * 8] = src[i];
#endif
        }
    }
    for (int i = tid; i < KN * FF; i += 128) cw[i / FF][i % FF] = conv_w[i];
    for (int i = tid; i < ROWS + KN - 1; i += 128) {
        int s = s0 + i - PADC;
        xh[i] = (s >= 0 && s < TE) ? cum[b * TE + s] : 0.f;   // SAME padding
    }
    {
        const float* q = dec + ((long)b * TD + t) * DD;
        for (int i = tid; i < DD; i += 128) { qvb[i] = q[i] + b_a[i]; va[i] = v_a[i]; }
    }
#if ASYNC_LDS
    wait_async0();
#endif
    __syncthreads();

    // Depthwise-style conv: channel = tid&31, rows strided by 4.
    {
        const int c = tid & 31;
        const float bias = conv_b[c];
        for (int r = tid >> 5; r < ROWS; r += 4) {
            float acc = bias;
            #pragma unroll
            for (int j = 0; j < KN; j++) acc += cw[j][c] * xh[r + j];
            fT[r][c] = (__bf16)acc;
        }
    }
    __syncthreads();

    // WMMA + fused tanh reduction. A-tile fixed per wave; loop over 32 N-tiles.
    const int m  = wv;
    const int hi = lane >> 4;     // lane half
    const int ln = lane & 15;

    const v16bf afrag = load16bf_split(&fT[m * 16 + ln][8 * hi]);

    float esum[8];
    #pragma unroll
    for (int r = 0; r < 8; r++) esum[r] = 0.f;

    const long encBase = ((long)b * TE + s0 + m * 16) * DD;

    for (int n = 0; n < DD / 16; n++) {
        const int d = n * 16 + ln;
        const v16bf bfrag = load16bf_split(&wl[d][8 * hi]);
        v8f c = {};
        c = __builtin_amdgcn_wmma_f32_16x16x32_bf16(
                false, afrag, false, bfrag, (short)0, c, false, false);
        const float qb = qvb[d];
        const float vv = va[d];
        const float* ep = enc + encBase + (long)(8 * hi) * DD + d;
        #pragma unroll
        for (int r = 0; r < 8; r++) {
            float x = ep[(long)r * DD] + qb + c[r];
            esum[r] += vv * fast_tanh(x);
        }
    }

    // Reduce the 16 d-phases (lanes within each half) per row.
    #pragma unroll
    for (int r = 0; r < 8; r++) {
        float v = esum[r];
        #pragma unroll
        for (int off = 1; off < 16; off <<= 1) v += __shfl_xor(v, off, 16);
        esum[r] = v;
    }
    if (ln == 0) {
        float* out = e_scr + b * TE + s0 + m * 16 + 8 * hi;
        #pragma unroll
        for (int r = 0; r < 8; r++) out[r] = esum[r];
    }
}

// ---------- K2: per-batch softmax over Te, emit f32 + bf16, update cumulative ----------

__launch_bounds__(256)
__global__ void k_softmax(const float* __restrict__ e_scr,
                          float* __restrict__ cum,
                          float* __restrict__ e_out,       // d_out e section
                          __bf16* __restrict__ ebf,        // bf16 copy for final GEMM
                          int t) {
    __shared__ float rmax[8];
    __shared__ float rsum[8];
    const int b = blockIdx.x, tid = threadIdx.x;
    const int lane = tid & 31, wv = tid >> 5;

    float v[4];
    float mx = -3.4e38f;
    #pragma unroll
    for (int i = 0; i < 4; i++) {
        v[i] = e_scr[b * TE + tid + 256 * i];
        mx = fmaxf(mx, v[i]);
    }
    #pragma unroll
    for (int off = 16; off >= 1; off >>= 1) mx = fmaxf(mx, __shfl_xor(mx, off, 32));
    if (lane == 0) rmax[wv] = mx;
    __syncthreads();
    float gmx = rmax[0];
    #pragma unroll
    for (int i = 1; i < 8; i++) gmx = fmaxf(gmx, rmax[i]);

    float s = 0.f;
    #pragma unroll
    for (int i = 0; i < 4; i++) { v[i] = __expf(v[i] - gmx); s += v[i]; }
    #pragma unroll
    for (int off = 16; off >= 1; off >>= 1) s += __shfl_xor(s, off, 32);
    if (lane == 0) rsum[wv] = s;
    __syncthreads();
    float tot = 0.f;
    #pragma unroll
    for (int i = 0; i < 8; i++) tot += rsum[i];
    const float inv = 1.f / tot;

    const long ebase = ((long)b * TD + t) * TE;
    #pragma unroll
    for (int i = 0; i < 4; i++) {
        const int si = tid + 256 * i;
        const float a = v[i] * inv;
        e_out[ebase + si] = a;
        ebf[ebase + si]   = (__bf16)a;
        cum[b * TE + si] += a;
    }
}

// ---------- K3: c[b] = e[b] (512x1024) @ enc[b] (1024x512), bf16 WMMA ----------
// grid (DD/16, TD/(16*8), B_), 256 threads (8 waves); each wave -> one 16x16 tile.
// B panel staged row-major (K x N) so each 32B row is two contiguous b128
// transfers (async-to-LDS when available).

__launch_bounds__(256)
__global__ void k_ctx(const __bf16* __restrict__ ebf,     // [B][TD][TE]
                      const __bf16* __restrict__ encb,    // [B][TE][DD]
                      float* __restrict__ cout) {         // [B][TD][DD]
    __shared__ __align__(16) __bf16 bs[32][16 + 8];        // B panel (K-major, padded)

    const int tid  = threadIdx.x;
    const int wv   = tid >> 5, lane = tid & 31;
    const int hi   = lane >> 4, ln = lane & 15;
    const int b    = blockIdx.z;
    const int n    = blockIdx.x;                 // N tile (D)
    const int mt   = blockIdx.y * 8 + wv;        // M tile (Td), one per wave

    const __bf16* Abase = ebf  + ((long)b * TD + mt * 16) * TE;
    const __bf16* Bbase = encb + (long)b * TE * DD + n * 16;

    v8f acc = {};
    for (int k = 0; k < TE / 32; k++) {
        // Cooperative stage of the 32x16 B panel + prefetch of the next panel.
        if (tid < 64) {
            const int row = tid >> 1, half = tid & 1;
            const __bf16* gp = Bbase + (long)(k * 32 + row) * DD + half * 8;
#if ASYNC_LDS
            async_copy_b128(gp, &bs[row][half * 8]);
#else
            *(uint4*)&bs[row][half * 8] = *(const uint4*)gp;
#endif
            if (k + 1 < TE / 32)
                __builtin_prefetch(gp + (long)32 * DD, 0, 1);   // global_prefetch_b8
        }
#if ASYNC_LDS
        wait_async0();
#endif
        __syncthreads();

        const v16bf af = load16bf_contig(Abase + (long)ln * TE + k * 32 + 16 * hi);
        v16bf bf;
        #pragma unroll
        for (int i = 0; i < 16; i++) bf[i] = bs[16 * hi + i][ln];
        acc = __builtin_amdgcn_wmma_f32_16x16x32_bf16(
                  false, af, false, bf, (short)0, acc, false, false);
        __syncthreads();
    }

    float* crow = cout + ((long)b * TD + mt * 16 + 8 * hi) * DD + n * 16 + ln;
    #pragma unroll
    for (int r = 0; r < 8; r++) crow[(long)r * DD] = acc[r];
}

// ---------- host launch ----------

extern "C" void kernel_launch(void* const* d_in, const int* in_sizes, int n_in,
                              void* d_out, int out_size, void* d_ws, size_t ws_size,
                              hipStream_t stream) {
    const float* enc   = (const float*)d_in[0];
    const float* dec   = (const float*)d_in[1];
    const float* convw = (const float*)d_in[2];
    const float* convb = (const float*)d_in[3];
    const float* wloc  = (const float*)d_in[4];
    const float* v_a   = (const float*)d_in[5];
    const float* b_a   = (const float*)d_in[6];

    float* c_out = (float*)d_out;                          // [B][TD][DD]
    float* e_out = c_out + (long)B_ * TD * DD;             // [B][TD][TE]

    char* ws = (char*)d_ws;
    float*  cum   = (float*)ws;  ws += (size_t)B_ * TE * 4;
    float*  e_scr = (float*)ws;  ws += (size_t)B_ * TE * 4;
    __bf16* wlocT = (__bf16*)ws; ws += (size_t)FF * DD * 2;
    __bf16* encb  = (__bf16*)ws; ws += (size_t)B_ * TE * DD * 2;
    __bf16* ebf   = (__bf16*)ws; ws += (size_t)B_ * TD * TE * 2;

    k_init<<<512, 256, 0, stream>>>(enc, wloc, encb, wlocT, cum);

    for (int t = 0; t < TD; t++) {
        dim3 g1(TE / ROWS, B_);
        k_step<<<g1, 128, 0, stream>>>(enc, dec, convw, convb, wlocT,
                                       v_a, b_a, cum, e_scr, t);
        k_softmax<<<B_, 256, 0, stream>>>(e_scr, cum, e_out, ebf, t);
    }

    dim3 g3(DD / 16, TD / (16 * 8), B_);
    k_ctx<<<g3, 256, 0, stream>>>(ebf, encb, c_out);
}
